// MultiHeadAttention_84533546320015
// MI455X (gfx1250) — matrix operand order repaired and checked by reading, not compiled, not run on hardware
//
#include <hip/hip_runtime.h>
#include <stdint.h>

// ---------------------------------------------------------------------------
// MI455X (gfx1250, wave32) multi-head attention, bf16 WMMA pipeline.
//   x,W (f32) --cvt--> bf16 (W transposed)  -> GEMM (WMMA bf16, f32 acc)
//   Q:[B,H,S,Hd]  K:[B,H,S,Hd]  V^T:[B,H,Hd,S]  (bf16)
//   flash-attention per (b,h): WMMA QK^T + online softmax + WMMA P*V
//   ctx:[B,S,H*Hd] bf16 -> GEMM with Wo^T -> f32 out + bias
// Tile staging uses the CDNA5 Tensor Data Mover (TENSOR_LOAD_TO_LDS, D# in
// SGPRs, TENSORcnt) with double-buffered LDS:
//   s_wait_tensorcnt 0 -> barrier -> (wave0 issues next tile's TDM) -> compute
// ---------------------------------------------------------------------------

typedef __attribute__((ext_vector_type(16))) __bf16    v16bf;
typedef __attribute__((ext_vector_type(8)))  __bf16    v8bf;
typedef __attribute__((ext_vector_type(8)))  float     v8f;
typedef __attribute__((ext_vector_type(4)))  unsigned  v4u;
typedef __attribute__((ext_vector_type(8)))  unsigned  v8u;

#define NB   4
#define NS   2048
#define ND   1024
#define NH   16
#define NHD  64
#define NM   (NB * NS)          // 8192 rows

static __device__ __forceinline__ v8f wmma_bf16(v16bf a, v16bf b, v8f c) {
  return __builtin_amdgcn_wmma_f32_16x16x32_bf16(
      /*neg_a=*/false, a, /*neg_b=*/false, b,
      /*c_mod=*/(short)0, c, /*reuse_a=*/false, /*reuse_b=*/false);
}

// --------------------- Tensor Data Mover (TDM) staging ---------------------
// 2D tile of a row-major 2-byte-element tensor -> contiguous LDS.
// D# group0/group1 per CDNA5 ISA ch.8; groups 2/3 NULL (2D tensor).
static __device__ __forceinline__ void tdm_load_2d(
    void* lds, const void* gmem,
    unsigned tensor_d0, unsigned tensor_d1, unsigned stride0,
    unsigned tile_d0, unsigned tile_d1) {
  const unsigned long long ga = (unsigned long long)(uintptr_t)gmem;
  v4u g0;
  g0[0] = 1u;                                    // count=1, user descriptor
  g0[1] = (unsigned)(uintptr_t)lds;              // lds_addr (bytes)
  g0[2] = (unsigned)ga;                          // global_addr[31:0]
  g0[3] = (unsigned)(ga >> 32) | 0x80000000u;    // global_addr[56:32] | type=2
  v8u g1;
  g1[0] = 1u << 16;                              // data_size=1 (2B); no mask/pad
  g1[1] = (tensor_d0 & 0xFFFFu) << 16;           // tensor_dim0[15:0]
  g1[2] = (tensor_d0 >> 16) | ((tensor_d1 & 0xFFFFu) << 16);  // dim0 hi | dim1 lo
  g1[3] = (tensor_d1 >> 16) | (tile_d0 << 16);   // dim1 hi | tile_dim0
  g1[4] = tile_d1;                               // tile_dim1 (tile_dim2 = 0)
  g1[5] = stride0;                               // tensor_dim0_stride[31:0]
  g1[6] = 0u;                                    // stride0 hi | dim1_stride lo
  g1[7] = 0u;
  asm volatile("tensor_load_to_lds %0, %1" :: "s"(g0), "s"(g1) : "memory");
}
static __device__ __forceinline__ void wait_tensor0() {
  __builtin_amdgcn_s_wait_tensorcnt(0);
}

// A fragment (16x32, bf16) from LDS tile laid out [row][K], row stride = `stride` elems.
// lane<16: row=lane,   K halves {0..7, 16..23}
// lane>=16: row=lane-16, K halves {8..15, 24..31}
static __device__ __forceinline__ v16bf load_frag_a(const __bf16* p, int stride) {
  const int lane = threadIdx.x & 31;
  const int row  = lane & 15;
  const int koff = (lane >> 4) << 3;                // 0 or 8
  const __bf16* r0 = p + row * stride + koff;
  v8bf lo = *(const v8bf*)(r0);                     // K: koff .. koff+7
  v8bf hi = *(const v8bf*)(r0 + 16);                // K: koff+16 .. koff+23
  return __builtin_shufflevector(lo, hi, 0,1,2,3,4,5,6,7,8,9,10,11,12,13,14,15);
}

// B fragment (32x16, bf16) from LDS tile holding B^T laid out [col][K].
// lane<16: col=lane, K {0..7, 16..23} ; lane>=16: col=lane-16, K {8..15, 24..31}
static __device__ __forceinline__ v16bf load_frag_b(const __bf16* p, int stride) {
  const int lane = threadIdx.x & 31;
  const int col  = lane & 15;
  const int koff = (lane >> 4) << 3;                // 0 or 8
  const __bf16* r0 = p + col * stride + koff;
  v8bf lo = *(const v8bf*)(r0);
  v8bf hi = *(const v8bf*)(r0 + 16);
  return __builtin_shufflevector(lo, hi, 0,1,2,3,4,5,6,7,8,9,10,11,12,13,14,15);
}

// --------------------------- conversion kernels ----------------------------

__global__ __launch_bounds__(256) void cvt_f32_bf16(const float* __restrict__ in,
                                                    __bf16* __restrict__ out, int n) {
  int i = blockIdx.x * 256 + threadIdx.x;
  if (i < n) out[i] = (__bf16)in[i];
}

// W [K=1024][N=1024] f32  ->  Wt [N][K] bf16 (LDS tiled transpose)
__global__ __launch_bounds__(1024) void cvt_w_transpose(const float* __restrict__ W,
                                                        __bf16* __restrict__ Wt) {
  __shared__ float t[32][33];
  const int kx = blockIdx.y * 32 + threadIdx.y;
  const int nx = blockIdx.x * 32 + threadIdx.x;
  t[threadIdx.y][threadIdx.x] = W[(size_t)kx * ND + nx];
  __syncthreads();
  const int n2 = blockIdx.x * 32 + threadIdx.y;
  const int k2 = blockIdx.y * 32 + threadIdx.x;
  Wt[(size_t)n2 * ND + k2] = (__bf16)t[threadIdx.x][threadIdx.y];
}

// ------------------------------- WMMA GEMM ---------------------------------
// C[M,N] = A[M,K] @ B[K,N] + bias ; A row-major bf16, Bt = B^T row-major bf16.
// Block 128x128x32, 256 threads = 8 waves (2x4), wave tile 64x32 (4x2 WMMA).
// Double-buffered LDS fed by TDM descriptors issued by wave 0.

enum { OUT_F32 = 0, OUT_QK = 1, OUT_VT = 2 };

template <int LAYOUT>
__global__ __launch_bounds__(256, 1) void gemm_bf16_wmma(
    const __bf16* __restrict__ A, const __bf16* __restrict__ Bt,
    const float* __restrict__ bias, void* __restrict__ Cout,
    int M, int N, int K) {
  constexpr int BM = 128, BN = 128, BK = 32;
  __shared__ alignas(16) __bf16 As[2][BM][BK];
  __shared__ alignas(16) __bf16 Bs[2][BN][BK];

  const int tid  = threadIdx.x;
  const int wave = tid >> 5;
  const int lane = tid & 31;
  const int wm   = wave >> 2;          // 0..1
  const int wn   = wave & 3;           // 0..3
  const int m0   = blockIdx.y * BM;
  const int n0   = blockIdx.x * BN;

  auto stage = [&](int buf, int k0) {
    if (wave == 0) {
      tdm_load_2d(&As[buf][0][0], A  + (size_t)m0 * K + k0, K, M, K, BK, BM);
      tdm_load_2d(&Bs[buf][0][0], Bt + (size_t)n0 * K + k0, K, N, K, BK, BN);
    }
  };

  v8f acc[4][2] = {};
  const int nk = K / BK;
  stage(0, 0);

  for (int kt = 0; kt < nk; ++kt) {
    wait_tensor0();                    // issuer's TDM done (others pass free)
    __syncthreads();                   // tile visible to all waves
    if (kt + 1 < nk) stage((kt + 1) & 1, (kt + 1) * BK);

    const int buf = kt & 1;
    v16bf af[4], bfr[2];
#pragma unroll
    for (int mt = 0; mt < 4; ++mt) af[mt] = load_frag_a(&As[buf][wm * 64 + mt * 16][0], BK);
#pragma unroll
    for (int nt = 0; nt < 2; ++nt) bfr[nt] = load_frag_b(&Bs[buf][wn * 32 + nt * 16][0], BK);
#pragma unroll
    for (int mt = 0; mt < 4; ++mt)
#pragma unroll
      for (int nt = 0; nt < 2; ++nt) acc[mt][nt] = wmma_bf16(af[mt], bfr[nt], acc[mt][nt]);
  }

  const int lh = lane >> 4;            // 0/1 -> row offset 0/8
  const int ln = lane & 15;            // output column within 16
#pragma unroll
  for (int mt = 0; mt < 4; ++mt) {
#pragma unroll
    for (int nt = 0; nt < 2; ++nt) {
      const int nG = n0 + wn * 32 + nt * 16 + ln;
      const float bb = bias[nG];
#pragma unroll
      for (int r = 0; r < 8; ++r) {
        const int mG = m0 + wm * 64 + mt * 16 + r + lh * 8;
        const float v = acc[mt][nt][r] + bb;
        if (LAYOUT == OUT_F32) {
          ((float*)Cout)[(size_t)mG * N + nG] = v;
        } else {
          const int b = mG >> 11, s = mG & (NS - 1);
          const int h = nG >> 6,  hd = nG & (NHD - 1);
          if (LAYOUT == OUT_QK)
            ((__bf16*)Cout)[((((size_t)b * NH + h) * NS) + s) * NHD + hd] = (__bf16)v;
          else  // OUT_VT : [B,H,Hd,S]
            ((__bf16*)Cout)[((((size_t)b * NH + h) * NHD) + hd) * NS + s] = (__bf16)v;
        }
      }
    }
  }
}

// --------------------------- flash attention -------------------------------
// Grid: (S/64, B*H). Block: 128 threads = 4 waves; wave owns 16 query rows.
// K/V tiles double-buffered, TDM staged (wave 0 issues descriptors).

__global__ __launch_bounds__(128, 1) void attn_fwd(
    const __bf16* __restrict__ Q,   // [B*H][S][Hd]
    const __bf16* __restrict__ Km,  // [B*H][S][Hd]
    const __bf16* __restrict__ Vt,  // [B*H][Hd][S]
    __bf16* __restrict__ O) {       // [B][S][H][Hd]
  constexpr float scale = 0.125f;   // 1/sqrt(64)
  __shared__ alignas(16) __bf16 qs[64][64];
  __shared__ alignas(16) __bf16 ps[64][64];      // P tile: [m][t]
  __shared__ alignas(16) __bf16 ks[2][64][64];   // K tile: [t][hd]
  __shared__ alignas(16) __bf16 vs[2][64][64];   // V^T tile: [hd][t]

  const int tid  = threadIdx.x;
  const int wave = tid >> 5;
  const int lane = tid & 31;
  const int lh   = lane >> 4;
  const int ln   = lane & 15;
  const int bh   = blockIdx.y;
  const int q0   = blockIdx.x * 64;

  const __bf16* Qb = Q  + (size_t)bh * NS * NHD;
  const __bf16* Kb = Km + (size_t)bh * NS * NHD;
  const __bf16* Vb = Vt + (size_t)bh * NHD * NS;

  auto stage_kv = [&](int buf, int t0) {
    if (wave == 0) {
      // K tile: rows t0..t0+63 of [S][Hd] -> ks[buf][t][hd]
      tdm_load_2d(&ks[buf][0][0], Kb + (size_t)t0 * NHD, NHD, NS, NHD, 64, 64);
      // V^T tile: cols t0..t0+63 of [Hd][S] -> vs[buf][hd][t]
      tdm_load_2d(&vs[buf][0][0], Vb + t0, NS, NHD, NS, 64, 64);
    }
  };

  if (wave == 0) {  // Q tile once
    tdm_load_2d(&qs[0][0], Qb + (size_t)q0 * NHD, NHD, NS, NHD, 64, 64);
  }
  stage_kv(0, 0);

  v8f acco[4] = {};
  float mrow[8], lrow[8];
#pragma unroll
  for (int r = 0; r < 8; ++r) { mrow[r] = -3.4e38f; lrow[r] = 0.f; }

  constexpr int NT = NS / 64;          // 32 key tiles
  for (int it = 0; it < NT; ++it) {
    wait_tensor0();
    __syncthreads();
    if (it + 1 < NT) stage_kv((it + 1) & 1, (it + 1) * 64);
    const int buf = it & 1;

    // S = Q @ K^T  (wave's 16 rows x 64 keys)
    v8f sacc[4] = {};
#pragma unroll
    for (int kk = 0; kk < 64; kk += 32) {
      v16bf aq = load_frag_a(&qs[wave * 16][kk], 64);
#pragma unroll
      for (int nt = 0; nt < 4; ++nt) {
        v16bf bk = load_frag_b(&ks[buf][nt * 16][kk], 64);
        sacc[nt] = wmma_bf16(aq, bk, sacc[nt]);
      }
    }

    // online softmax (per row; rows live per-VGPR, 16-lane halves share a row)
    float pv[4][8];
    float alpha[8];
#pragma unroll
    for (int r = 0; r < 8; ++r) {
      float sv[4];
      float mx = -3.4e38f;
#pragma unroll
      for (int nt = 0; nt < 4; ++nt) { sv[nt] = sacc[nt][r] * scale; mx = fmaxf(mx, sv[nt]); }
#pragma unroll
      for (int off = 1; off < 16; off <<= 1) mx = fmaxf(mx, __shfl_xor(mx, off, 32));
      const float mnew = fmaxf(mrow[r], mx);
      const float a = __expf(mrow[r] - mnew);
      float s = 0.f;
#pragma unroll
      for (int nt = 0; nt < 4; ++nt) { float p = __expf(sv[nt] - mnew); pv[nt][r] = p; s += p; }
#pragma unroll
      for (int off = 1; off < 16; off <<= 1) s += __shfl_xor(s, off, 32);
      lrow[r] = lrow[r] * a + s;
      mrow[r] = mnew;
      alpha[r] = a;
    }
#pragma unroll
    for (int nt = 0; nt < 4; ++nt)
#pragma unroll
      for (int r = 0; r < 8; ++r) acco[nt][r] *= alpha[r];

    // P -> LDS (bf16) so it can be re-read in A-fragment layout (wave-local, DS in-order)
#pragma unroll
    for (int nt = 0; nt < 4; ++nt)
#pragma unroll
      for (int r = 0; r < 8; ++r)
        ps[wave * 16 + r + lh * 8][nt * 16 + ln] = (__bf16)pv[nt][r];

    // ctx += P @ V
#pragma unroll
    for (int kk = 0; kk < 64; kk += 32) {
      v16bf ap = load_frag_a(&ps[wave * 16][kk], 64);
#pragma unroll
      for (int nt = 0; nt < 4; ++nt) {
        v16bf bv = load_frag_b(&vs[buf][nt * 16][kk], 64);
        acco[nt] = wmma_bf16(ap, bv, acco[nt]);
      }
    }
  }

  // write ctx / l : O[b][s][h][hd] bf16
  const int b = bh >> 4, h = bh & 15;
#pragma unroll
  for (int r = 0; r < 8; ++r) {
    const int s = q0 + wave * 16 + r + lh * 8;
    const float inv = 1.0f / lrow[r];
#pragma unroll
    for (int nt = 0; nt < 4; ++nt) {
      const int hd = nt * 16 + ln;
      O[(((size_t)b * NS + s) * NH + h) * NHD + hd] = (__bf16)(acco[nt][r] * inv);
    }
  }
}

// ------------------------------- launcher ----------------------------------

extern "C" void kernel_launch(void* const* d_in, const int* in_sizes, int n_in,
                              void* d_out, int out_size, void* d_ws, size_t ws_size,
                              hipStream_t stream) {
  (void)in_sizes; (void)n_in; (void)out_size; (void)ws_size;
  const float* x  = (const float*)d_in[0];
  const float* Wq = (const float*)d_in[1];
  const float* bq = (const float*)d_in[2];
  const float* Wk = (const float*)d_in[3];
  const float* bk = (const float*)d_in[4];
  const float* Wv = (const float*)d_in[5];
  const float* bv = (const float*)d_in[6];
  const float* Wo = (const float*)d_in[7];
  const float* bo = (const float*)d_in[8];

  char* w = (char*)d_ws;
  __bf16* xb  = (__bf16*)w;  w += (size_t)NM * ND * 2;        // 16 MB
  __bf16* wqt = (__bf16*)w;  w += (size_t)ND * ND * 2;        //  2 MB
  __bf16* wkt = (__bf16*)w;  w += (size_t)ND * ND * 2;
  __bf16* wvt = (__bf16*)w;  w += (size_t)ND * ND * 2;
  __bf16* wot = (__bf16*)w;  w += (size_t)ND * ND * 2;
  __bf16* qws = (__bf16*)w;  w += (size_t)NM * ND * 2;        // 16 MB
  __bf16* kws = (__bf16*)w;  w += (size_t)NM * ND * 2;
  __bf16* vws = (__bf16*)w;  w += (size_t)NM * ND * 2;
  __bf16* ctx = (__bf16*)w;  w += (size_t)NM * ND * 2;

  const int nX = NM * ND;
  cvt_f32_bf16<<<(nX + 255) / 256, 256, 0, stream>>>(x, xb, nX);
  dim3 tg(32, 32), tb(32, 32);
  cvt_w_transpose<<<tg, tb, 0, stream>>>(Wq, wqt);
  cvt_w_transpose<<<tg, tb, 0, stream>>>(Wk, wkt);
  cvt_w_transpose<<<tg, tb, 0, stream>>>(Wv, wvt);
  cvt_w_transpose<<<tg, tb, 0, stream>>>(Wo, wot);

  dim3 gg(ND / 128, NM / 128);  // (8, 64)
  gemm_bf16_wmma<OUT_QK><<<gg, 256, 0, stream>>>(xb, wqt, bq, qws, NM, ND, ND);
  gemm_bf16_wmma<OUT_QK><<<gg, 256, 0, stream>>>(xb, wkt, bk, kws, NM, ND, ND);
  gemm_bf16_wmma<OUT_VT><<<gg, 256, 0, stream>>>(xb, wvt, bv, vws, NM, ND, ND);

  dim3 ag(NS / 64, NB * NH);    // (32, 64)
  attn_fwd<<<ag, 128, 0, stream>>>(qws, kws, vws, ctx);

  gemm_bf16_wmma<OUT_F32><<<gg, 256, 0, stream>>>(ctx, wot, bo, d_out, NM, ND, ND);
}
